// DGM_d_21998822490117
// MI455X (gfx1250) — matrix operand order repaired and checked by reading, not compiled, not run on hardware
//
#include <hip/hip_runtime.h>
#include <hip/hip_bf16.h>
#include <stdint.h>

// ---------------------------------------------------------------------------
// Problem constants (from reference): B=4, N=4096, D_IN=128, D_EMB=64, K=16
// ---------------------------------------------------------------------------
#define BB 4
#define NN 4096
#define DIN 128
#define DE 64
#define KK 16

typedef __attribute__((ext_vector_type(8)))  float          v8f;
typedef __attribute__((ext_vector_type(16))) __bf16         v16bf;
typedef __attribute__((ext_vector_type(8)))  unsigned short ushort8;
typedef __attribute__((ext_vector_type(4)))  float          f4;

struct U16Pair { ushort8 a; ushort8 b; };

__device__ __forceinline__ v16bf make_frag(ushort8 x, ushort8 y) {
    U16Pair p{x, y};
    return __builtin_bit_cast(v16bf, p);
}

// Raw hardware log2 (v_log_f32), no denormal fixup / extended-precision ln path.
__device__ __forceinline__ float fast_log2(float x) {
#if __has_builtin(__builtin_amdgcn_logf)
    return __builtin_amdgcn_logf(x);
#else
    float r;
    asm volatile("v_log_f32 %0, %1" : "=v"(r) : "v"(x));
    return r;
#endif
}

// A-operand (16-bit A 16x32, ISA 7.12.2): lane<16 -> M=lane, K chunks at 0..7 / 16..23
//                                         lane>=16 -> M=lane-16, K chunks at 8..15 / 24..31
__device__ __forceinline__ v16bf load_fragA(const unsigned short* __restrict__ p,
                                            int baseRow, int lane, int khalf) {
    size_t off = (size_t)(baseRow + (lane & 15)) * DE + ((lane >> 4) << 3) + (khalf << 5);
    ushort8 x = *(const ushort8*)(p + off);
    ushort8 y = *(const ushort8*)(p + off + 16);
    return make_frag(x, y);
}

// B-operand (16-bit B 32x16): lane<16 -> N=lane, K chunks at 0..7 / 16..23; lane>=16 -> N=lane-16, K chunks at 8..15 / 24..31.
// B column n == xe row (baseCol+n): two 8-half chunks per lane -> two B128 loads.
__device__ __forceinline__ v16bf load_fragB(const unsigned short* __restrict__ p,
                                            int baseCol, int lane, int khalf) {
    size_t off = (size_t)(baseCol + (lane & 15)) * DE + ((lane >> 4) << 3) + (khalf << 5);
    ushort8 x = *(const ushort8*)(p + off);
    ushort8 y = *(const ushort8*)(p + off + 16);
    return make_frag(x, y);
}

__device__ __forceinline__ v8f wmma_bf16(v16bf a, v16bf b, v8f c) {
    // (neg_a, A, neg_b, B, c_mod, C, reuse_a, reuse_b)
    return __builtin_amdgcn_wmma_f32_16x16x32_bf16(false, a, false, b, (short)0, c,
                                                   false, false);
}

__device__ __forceinline__ unsigned short f32_to_bf16_rne(float f) {
    uint32_t u = __float_as_uint(f);
    uint32_t r = u + 0x7FFFu + ((u >> 16) & 1u);
    return (unsigned short)(r >> 16);
}

// ---------------------------------------------------------------------------
// Kernel 1: xe = x@W + bias (fp32), plus bf16 hi/lo split of xe and row norms.
// One block = 4 rows x 64 cols; W cached in LDS (32KB).
// ---------------------------------------------------------------------------
__global__ __launch_bounds__(256) void dgm_embed_kernel(
    const float* __restrict__ x, const float* __restrict__ W,
    const float* __restrict__ bias, float* __restrict__ xeOut,
    unsigned short* __restrict__ xeHi, unsigned short* __restrict__ xeLo,
    float* __restrict__ sqArr)
{
    __shared__ float sW[DIN * DE];     // 32 KB
    __shared__ float sX[4 * DIN];      // 2 KB
    __shared__ float sRed[256];

    const int tid = threadIdx.x;
    const long rowBase = (long)blockIdx.x * 4;   // flat row index b*N + n

    { // cooperative LDS fills
        const f4* wv = (const f4*)W; f4* swv = (f4*)sW;
        #pragma unroll
        for (int i = 0; i < (DIN * DE / 4) / 256; ++i) swv[tid + i * 256] = wv[tid + i * 256];
        const f4* xv = (const f4*)(x + rowBase * DIN); f4* sxv = (f4*)sX;
        if (tid < 128) sxv[tid] = xv[tid];
    }
    __syncthreads();

    const int r = tid >> 6;        // 0..3
    const int e = tid & 63;        // 0..63
    float acc = bias[e];
    #pragma unroll 16
    for (int d = 0; d < DIN; ++d) acc = fmaf(sX[r * DIN + d], sW[d * DE + e], acc);

    const long o = (rowBase + r) * DE + e;
    xeOut[o] = acc;

    // hi/lo bf16 split: acc ~= hi + lo with both representable in bf16
    unsigned short hi = f32_to_bf16_rne(acc);
    float fhi = __uint_as_float(((uint32_t)hi) << 16);
    unsigned short lo = f32_to_bf16_rne(acc - fhi);
    xeHi[o] = hi;
    xeLo[o] = lo;

    // row squared-norm (exact fp32 xe)
    sRed[tid] = acc * acc;
    __syncthreads();
    if (e == 0) {
        float s = 0.f;
        #pragma unroll
        for (int d = 0; d < DE; ++d) s += sRed[r * DE + d];
        sqArr[rowBase + r] = s;
    }
}

// ---------------------------------------------------------------------------
// Kernel 2: fused gram (WMMA bf16 hi/lo) + distances + Gumbel noise + top-16.
// Block = 128 threads (4 waves); each wave owns 32 rows (two 16x16 subtiles)
// and iterates over 256 column tiles. Lane L owns full row L of the wave tile.
// ---------------------------------------------------------------------------
__global__ __launch_bounds__(128) void dgm_topk_kernel(
    const unsigned short* __restrict__ xeHi,
    const unsigned short* __restrict__ xeLo,
    const float* __restrict__ sqArr,
    const float* __restrict__ q,
    const float* __restrict__ temperature,
    float* __restrict__ out)
{
    __shared__ float sSq[NN];            // 16 KB: this batch's row norms
    __shared__ float sT[4][32 * 20];     // per-wave padded transpose scratch (10 KB)

    const int tid  = threadIdx.x;
    const int wave = tid >> 5;
    const int lane = tid & 31;
    const int bid  = blockIdx.x;
    const int b        = bid >> 5;            // 32 blocks per batch
    const int rowBlock = (bid & 31) * 128;    // first row (in batch) of this block

    { // cooperative load of sq for this batch
        const f4* sv = (const f4*)(sqArr + (size_t)b * NN);
        f4* dv = (f4*)sSq;
        #pragma unroll
        for (int i = 0; i < (NN / 4) / 128; ++i) dv[tid + i * 128] = sv[tid + i * 128];
    }
    __syncthreads();

    const float tclip = fminf(fmaxf(temperature[0], -5.f), 5.f);
    const float scale = __expf(tclip);

    const int  waveRow0 = rowBlock + wave * 32;        // in-batch row of wave tile
    const int  rowInB   = waveRow0 + lane;             // this lane's row
    const long gRow     = (long)b * NN + rowInB;       // global flat row
    const float sqRow   = sSq[rowInB];
    const int  aBase    = b * NN + waveRow0;           // flat row base for A frags

    // A fragments: 2 subtiles x {hi,lo} x 2 K-halves -> 64 VGPRs, resident.
    const v16bf ah0s0 = load_fragA(xeHi, aBase,      lane, 0);
    const v16bf ah1s0 = load_fragA(xeHi, aBase,      lane, 1);
    const v16bf al0s0 = load_fragA(xeLo, aBase,      lane, 0);
    const v16bf al1s0 = load_fragA(xeLo, aBase,      lane, 1);
    const v16bf ah0s1 = load_fragA(xeHi, aBase + 16, lane, 0);
    const v16bf ah1s1 = load_fragA(xeHi, aBase + 16, lane, 1);
    const v16bf al0s1 = load_fragA(xeLo, aBase + 16, lane, 0);
    const v16bf al1s1 = load_fragA(xeLo, aBase + 16, lane, 1);

    // Register-resident top-16 (min-replacement scheme)
    float tk[KK]; int ti[KK];
    #pragma unroll
    for (int s = 0; s < KK; ++s) { tk[s] = -3.4e38f; ti[s] = 0; }
    float cmin = -3.4e38f; int cslot = 0;

    float* __restrict__ myT = &sT[wave][0];
    const float* __restrict__ qRow = q + gRow * NN;
    const int cCol = lane & 15, rOff = (lane >> 4) * 8;

    const float LN2   = 0.69314718055994531f;   // ln(2)
    const float LNLN2 = -0.36651292058166433f;  // ln(ln(2))

    for (int j = 0; j < NN / 16; ++j) {
        const int jb = j * 16;
        const int bBase = b * NN + jb;

        // issue q loads early: latency hides under the 12 WMMAs below
        float qv[16];
        #pragma unroll
        for (int e4 = 0; e4 < 16; e4 += 4) {
            f4 t = *(const f4*)(qRow + jb + e4);
            qv[e4] = t[0]; qv[e4 + 1] = t[1]; qv[e4 + 2] = t[2]; qv[e4 + 3] = t[3];
        }
        __builtin_prefetch(qRow + jb + 64, 0, 0);   // 4 iterations ahead in lane's q row

        // B fragments from L2-resident bf16 xe copies
        v16bf bh0 = load_fragB(xeHi, bBase, lane, 0);
        v16bf bh1 = load_fragB(xeHi, bBase, lane, 1);
        v16bf bl0 = load_fragB(xeLo, bBase, lane, 0);
        v16bf bl1 = load_fragB(xeLo, bBase, lane, 1);

        // gram tiles: hi*hi + hi*lo + lo*hi  (~fp32 accuracy)
        v8f c0 = {0.f, 0.f, 0.f, 0.f, 0.f, 0.f, 0.f, 0.f};
        c0 = wmma_bf16(ah0s0, bh0, c0); c0 = wmma_bf16(ah1s0, bh1, c0);
        c0 = wmma_bf16(ah0s0, bl0, c0); c0 = wmma_bf16(ah1s0, bl1, c0);
        c0 = wmma_bf16(al0s0, bh0, c0); c0 = wmma_bf16(al1s0, bh1, c0);
        v8f c1 = {0.f, 0.f, 0.f, 0.f, 0.f, 0.f, 0.f, 0.f};
        c1 = wmma_bf16(ah0s1, bh0, c1); c1 = wmma_bf16(ah1s1, bh1, c1);
        c1 = wmma_bf16(ah0s1, bl0, c1); c1 = wmma_bf16(ah1s1, bl1, c1);
        c1 = wmma_bf16(al0s1, bh0, c1); c1 = wmma_bf16(al1s1, bh1, c1);

        // transpose C tiles through padded LDS so each lane owns a full row
        #pragma unroll
        for (int v = 0; v < 8; ++v) myT[(rOff + v) * 20 + cCol] = c0[v];
        #pragma unroll
        for (int v = 0; v < 8; ++v) myT[(16 + rOff + v) * 20 + cCol] = c1[v];
        __builtin_amdgcn_wave_barrier();
        asm volatile("s_wait_dscnt 0" ::: "memory");   // same-wave LDS RAW

        float g[16];
        #pragma unroll
        for (int e4 = 0; e4 < 16; e4 += 4) {
            f4 t = *(const f4*)&myT[lane * 20 + e4];
            g[e4] = t[0]; g[e4 + 1] = t[1]; g[e4 + 2] = t[2]; g[e4 + 3] = t[3];
        }

        #pragma unroll
        for (int e = 0; e < 16; ++e) {
            float dv = fmaxf(sqRow + sSq[jb + e] - 2.f * g[e], 0.f);
            // log(-log(q+eps)) == ln2*log2(-log2(q+eps)) + ln(ln2): two raw v_log_f32
            float l1  = fast_log2(qv[e] + 1e-8f);
            float key = fmaf(LN2, fast_log2(-l1), LNLN2) - dv * scale; // = -lq
            if (key > cmin) {
                const int idx = jb + e;
                #pragma unroll
                for (int s = 0; s < KK; ++s)
                    if (s == cslot) { tk[s] = key; ti[s] = idx; }
                cmin = tk[0]; cslot = 0;
                #pragma unroll
                for (int s = 1; s < KK; ++s)
                    if (tk[s] < cmin) { cmin = tk[s]; cslot = s; }
            }
        }
        __builtin_amdgcn_wave_barrier();   // keep next j's stores behind our loads
    }

    // sort the 16 survivors descending (lax.top_k order)
    #pragma unroll
    for (int a = 0; a < KK - 1; ++a)
        #pragma unroll
        for (int c2 = a + 1; c2 < KK; ++c2)
            if (tk[c2] > tk[a]) {
                float tf = tk[a]; tk[a] = tk[c2]; tk[c2] = tf;
                int   tj = ti[a]; ti[a] = ti[c2]; ti[c2] = tj;
            }

    // outputs: [xe | edges(2 x B*N*K) | logprobs(B*N*K)]
    const long xeSize = (long)BB * NN * DE;
    const long E      = (long)BB * NN * KK;
    float* e0 = out + xeSize;
    float* e1 = out + xeSize + E;
    float* lp = out + xeSize + 2 * E;
    const long p = gRow * KK;
    const float rowEdge = (float)(rowInB + b * NN);
    #pragma unroll
    for (int k = 0; k < KK; ++k) {
        e0[p + k] = (float)(ti[k] + b * NN);
        e1[p + k] = rowEdge;
        lp[p + k] = tk[k];
    }
}

// ---------------------------------------------------------------------------
extern "C" void kernel_launch(void* const* d_in, const int* in_sizes, int n_in,
                              void* d_out, int out_size, void* d_ws, size_t ws_size,
                              hipStream_t stream) {
    const float* x    = (const float*)d_in[0];
    // d_in[1] = A (unused placeholder)
    const float* W    = (const float*)d_in[2];
    const float* bias = (const float*)d_in[3];
    const float* temp = (const float*)d_in[4];
    const float* q    = (const float*)d_in[5];
    float* out = (float*)d_out;

    // workspace: xeHi (2MB bf16) | xeLo (2MB bf16) | sq (64KB f32)
    unsigned short* xeHi = (unsigned short*)d_ws;
    unsigned short* xeLo = xeHi + (size_t)BB * NN * DE;
    float*          sqA  = (float*)(xeLo + (size_t)BB * NN * DE);

    dgm_embed_kernel<<<BB * NN / 4, 256, 0, stream>>>(x, W, bias, out, xeHi, xeLo, sqA);
    dgm_topk_kernel<<<BB * NN / 128, 128, 0, stream>>>(xeHi, xeLo, sqA, q, temp, out);
}
